// Naive_30004641530481
// MI455X (gfx1250) — matrix operand order repaired and checked by reading, not compiled, not run on hardware
//
#include <hip/hip_runtime.h>

// ---------------------------------------------------------------------------
// Per-model 4-layer MLP (64x64), fused on-chip for MI455X (gfx1250, wave32).
// bf16x3 split-precision WMMA: x@W ~= xh@Wh + xh@Wl + xl@Wh, fp32 accumulate.
// Activations & weights live in LDS as separate bf16 hi/lo planes with padded
// row stride so WMMA A/B fragments are direct bank-friendly ds_read_b128s.
// ---------------------------------------------------------------------------

#define N_TOKENS   2097152
#define D          64
#define N_MODELS   64
#define N_LAYERS   4
#define TOK_PER_MODEL (N_TOKENS / N_MODELS)   // 32768

#define WAVES      8
#define TOKW       32                          // tokens per wave per iteration
#define TOK_BLK    (WAVES * TOKW)              // 256
#define CHUNKS     16                          // grid.y
#define TOK_CHUNK  (TOK_PER_MODEL / CHUNKS)    // 2048
#define ITERS      (TOK_CHUNK / TOK_BLK)       // 8

#define WSTR       72   // weight plane row stride (bf16 elems), 144 B
#define ASTR       72   // activation plane row stride (bf16 elems), 144 B
#define ACT_WAVE_U32 2304   // per-wave region: 2 planes * 32*72 bf16 = 9216 B

typedef __attribute__((ext_vector_type(16))) __bf16 v16bf;
typedef __attribute__((ext_vector_type(4)))  __bf16 v4bf;
typedef __attribute__((ext_vector_type(8)))  float  v8f;

union ABfrag { v16bf v; uint4 q[2]; };

__device__ __forceinline__ void split2(float f, __bf16& h, __bf16& l) {
    h = (__bf16)f;                 // RNE truncate to bf16
    l = (__bf16)(f - (float)h);    // residual in bf16
}

__global__ __launch_bounds__(256)
void mlp64_bf16x3_kernel(const float* __restrict__ x,
                         const float* __restrict__ W,
                         const float* __restrict__ bias,
                         float* __restrict__ out) {
    __shared__ __bf16   sWhi[N_LAYERS * D * WSTR];   // 36 KB
    __shared__ __bf16   sWlo[N_LAYERS * D * WSTR];   // 36 KB
    __shared__ float    sBias[N_LAYERS * D];         //  1 KB
    __shared__ unsigned sAct[WAVES * ACT_WAVE_U32];  // 72 KB

    const int tid   = threadIdx.x;
    const int wave  = tid >> 5;
    const int lane  = tid & 31;
    const int model = blockIdx.x;
    const int chunk = blockIdx.y;

    // ---- Stage W[model] into padded LDS planes, fp32 -> bf16 hi + lo ------
    {
        const float4* wsrc = (const float4*)(W + (size_t)model * (N_LAYERS * D * D));
        #pragma unroll 4
        for (int i = tid; i < (N_LAYERS * D * D) / 4; i += 256) {
            float4 v = wsrc[i];
            const int row = i >> 4;          // [layer*64 + f_out]
            const int col = (i & 15) * 4;    // d_in
            __bf16 h0, l0, h1, l1, h2, l2, h3, l3;
            split2(v.x, h0, l0); split2(v.y, h1, l1);
            split2(v.z, h2, l2); split2(v.w, h3, l3);
            v4bf h4 = {h0, h1, h2, h3};
            v4bf l4 = {l0, l1, l2, l3};
            *(v4bf*)&sWhi[row * WSTR + col] = h4;
            *(v4bf*)&sWlo[row * WSTR + col] = l4;
        }
        const float* bsrc = bias + (size_t)model * (N_LAYERS * D);
        if (tid < N_LAYERS * D) sBias[tid] = bsrc[tid];
    }
    __syncthreads();

    // Per-wave activation planes (hi, lo) and fp32 staging view (overlapped).
    unsigned* stage = &sAct[wave * ACT_WAVE_U32];
    __bf16*   hiP   = (__bf16*)stage;
    __bf16*   loP   = hiP + TOKW * ASTR;

    const size_t tokBase = (size_t)model * TOK_PER_MODEL + (size_t)chunk * TOK_CHUNK;

    for (int it = 0; it < ITERS; ++it) {
        const size_t rowBase = tokBase + (size_t)it * TOK_BLK + (size_t)wave * TOKW;

        // ---- Load x tile (coalesced b128), split into hi/lo planes --------
        {
            const float4* xsrc = (const float4*)(x + rowBase * D);
            #pragma unroll
            for (int i = 0; i < 16; ++i) {
                const int idx = i * 32 + lane;     // float4 idx in 32x64 tile
                const int row = idx >> 4;
                const int col = (idx & 15) * 4;
                float4 v = xsrc[idx];
                __bf16 h0, l0, h1, l1, h2, l2, h3, l3;
                split2(v.x, h0, l0); split2(v.y, h1, l1);
                split2(v.z, h2, l2); split2(v.w, h3, l3);
                v4bf h4 = {h0, h1, h2, h3};
                v4bf l4 = {l0, l1, l2, l3};
                *(v4bf*)&hiP[row * ASTR + col] = h4;
                *(v4bf*)&loP[row * ASTR + col] = l4;
            }
        }

        // ---- 4 fused layers ----------------------------------------------
        for (int l = 0; l < N_LAYERS; ++l) {
            v8f acc[2][4];
            #pragma unroll
            for (int rt = 0; rt < 2; ++rt)
                #pragma unroll
                for (int nt = 0; nt < 4; ++nt)
                    acc[rt][nt] = (v8f){0.f, 0.f, 0.f, 0.f, 0.f, 0.f, 0.f, 0.f};

            #pragma unroll
            for (int kt = 0; kt < 2; ++kt) {
                // A fragments (16x32 bf16): contiguous pair-packed reads.
                // lane<16: row=lane, K = k0..k0+7 and k0+16..k0+23
                // lane>=16: same row, K shifted by +8 (ISA A layout).
                ABfrag Ah[2], Al[2];
                #pragma unroll
                for (int rt = 0; rt < 2; ++rt) {
                    const int row = rt * 16 + (lane & 15);
                    const int k0  = kt * 32 + ((lane >> 4) ? 8 : 0);
                    const __bf16* ah = hiP + row * ASTR + k0;
                    const __bf16* al = loP + row * ASTR + k0;
                    Ah[rt].q[0] = *(const uint4*)(ah);
                    Ah[rt].q[1] = *(const uint4*)(ah + 16);
                    Al[rt].q[0] = *(const uint4*)(al);
                    Al[rt].q[1] = *(const uint4*)(al + 16);
                }
                #pragma unroll
                for (int nt = 0; nt < 4; ++nt) {
                    // B fragment: B[k][n] = W[n][k]; same K split as A:
                    // K = kb..kb+7 and kb+16..kb+23 (lane>=16: +8).
                    const int f  = nt * 16 + (lane & 15);
                    const int kb = kt * 32 + ((lane >> 4) ? 8 : 0);
                    const __bf16* bh = &sWhi[(l * D + f) * WSTR + kb];
                    const __bf16* bl = &sWlo[(l * D + f) * WSTR + kb];
                    ABfrag Bh, Bl;
                    Bh.q[0] = *(const uint4*)(bh);
                    Bh.q[1] = *(const uint4*)(bh + 16);
                    Bl.q[0] = *(const uint4*)(bl);
                    Bl.q[1] = *(const uint4*)(bl + 16);
                    #pragma unroll
                    for (int rt = 0; rt < 2; ++rt) {
                        acc[rt][nt] = __builtin_amdgcn_wmma_f32_16x16x32_bf16(
                            false, Ah[rt].v, false, Bh.v, (short)0, acc[rt][nt], false, false);
                        acc[rt][nt] = __builtin_amdgcn_wmma_f32_16x16x32_bf16(
                            false, Ah[rt].v, false, Bl.v, (short)0, acc[rt][nt], false, false);
                        acc[rt][nt] = __builtin_amdgcn_wmma_f32_16x16x32_bf16(
                            false, Al[rt].v, false, Bh.v, (short)0, acc[rt][nt], false, false);
                    }
                }
            }

            // ---- bias + ReLU epilogue -------------------------------------
            #pragma unroll
            for (int nt = 0; nt < 4; ++nt) {
                const float bv  = sBias[l * D + nt * 16 + (lane & 15)];
                const int   col = nt * 16 + (lane & 15);
                #pragma unroll
                for (int rt = 0; rt < 2; ++rt) {
                    #pragma unroll
                    for (int v = 0; v < 8; ++v) {
                        float y = fmaxf(acc[rt][nt][v] + bv, 0.0f);
                        const int row = rt * 16 + v + ((lane >> 4) ? 8 : 0);
                        if (l < N_LAYERS - 1) {
                            __bf16 h, lo;
                            split2(y, h, lo);
                            hiP[row * ASTR + col] = h;
                            loP[row * ASTR + col] = lo;
                        } else {
                            stage[row * 64 + col] = __float_as_uint(y);
                        }
                    }
                }
            }
        }

        // ---- Coalesced b128 store of the f32 result tile ------------------
        {
            uint4* dst = (uint4*)(out + rowBase * D);
            #pragma unroll
            for (int i = 0; i < 16; ++i) {
                const int idx = i * 32 + lane;
                dst[idx] = ((const uint4*)stage)[idx];
            }
        }
    }
}

extern "C" void kernel_launch(void* const* d_in, const int* in_sizes, int n_in,
                              void* d_out, int out_size, void* d_ws, size_t ws_size,
                              hipStream_t stream) {
    (void)in_sizes; (void)n_in; (void)d_ws; (void)ws_size; (void)out_size;
    const float* x = (const float*)d_in[0];
    const float* W = (const float*)d_in[1];
    const float* b = (const float*)d_in[2];
    float* out     = (float*)d_out;

    dim3 grid(N_MODELS, CHUNKS, 1);
    mlp64_bf16x3_kernel<<<grid, 256, 0, stream>>>(x, W, b, out);
}
